// RandomProjectionQuantizer_68367289418161
// MI455X (gfx1250) — matrix operand order repaired and checked by reading, not compiled, not run on hardware
//
#include <hip/hip_runtime.h>
#include <hip/hip_bf16.h>
#include <math.h>

// Problem constants (from reference): B=4, N=4096, D=1024, E=512, K=4096
#define M_ROWS   16384   // B*N
#define D_DIM    1024
#define E_DIM    512
#define K_CB     4096

typedef __attribute__((ext_vector_type(16))) __bf16 v16bf;
typedef __attribute__((ext_vector_type(8)))  __bf16 v8bf;
typedef __attribute__((ext_vector_type(8)))  float  v8f;

#if defined(__has_builtin)
#  if __has_builtin(__builtin_amdgcn_global_load_async_to_lds_b128)
#    define HAVE_ASYNC_LDS 1
#  endif
#endif
#ifndef HAVE_ASYNC_LDS
#  define HAVE_ASYNC_LDS 0
#endif

#if HAVE_ASYNC_LDS
// Parameter types per hipcc diagnostic: pointee is int __vector(4), non-const;
// src in global (AS1), dst in LDS (AS3).
typedef int v4i_vs __attribute__((vector_size(16)));
typedef __attribute__((address_space(1))) v4i_vs as1_v4i;
typedef __attribute__((address_space(3))) v4i_vs as3_v4i;
#endif

// fp32 -> bf16 via native convert (v_cvt_pk_bf16_f32 on gfx1250)
static __device__ __forceinline__ unsigned short f2bf(float f) {
    union { __bf16 h; unsigned short s; } u;
    u.h = (__bf16)f;
    return u.s;
}

static __device__ __forceinline__ v8f v8f_zero() {
    v8f z;
#pragma unroll
    for (int i = 0; i < 8; ++i) z[i] = 0.0f;
    return z;
}

#if HAVE_ASYNC_LDS
static __device__ __forceinline__ void wait_async0() {
#if __has_builtin(__builtin_amdgcn_s_wait_asynccnt)
    __builtin_amdgcn_s_wait_asynccnt(0);
#else
    asm volatile("s_wait_asynccnt 0x0" ::: "memory");
#endif
}
#endif

// ---------------------------------------------------------------------------
// Kernel: L2-normalize 512-float rows -> bf16. One block (128 thr) per row.
// ---------------------------------------------------------------------------
__global__ void k_rownorm_bf16(const float* __restrict__ in,
                               unsigned short* __restrict__ out) {
    __shared__ float red[128];
    const int row = blockIdx.x;
    const int tid = threadIdx.x;
    const float4 v = ((const float4*)(in + (size_t)row * E_DIM))[tid];
    red[tid] = v.x * v.x + v.y * v.y + v.z * v.z + v.w * v.w;
    __syncthreads();
    for (int st = 64; st > 0; st >>= 1) {
        if (tid < st) red[tid] += red[tid + st];
        __syncthreads();
    }
    const float scale = 1.0f / fmaxf(sqrtf(red[0]), 1e-12f);
    unsigned long long pk =
          (unsigned long long)f2bf(v.x * scale)
        | ((unsigned long long)f2bf(v.y * scale) << 16)
        | ((unsigned long long)f2bf(v.z * scale) << 32)
        | ((unsigned long long)f2bf(v.w * scale) << 48);
    ((unsigned long long*)(out + (size_t)row * E_DIM))[tid] = pk;
}

// ---------------------------------------------------------------------------
// Kernel: transpose P (D x E fp32) -> Pt (E x D bf16), LDS-tiled 32x32.
// ---------------------------------------------------------------------------
__global__ void k_transpose_bf16(const float* __restrict__ P,
                                 unsigned short* __restrict__ Pt) {
    __shared__ float tile[32][33];
    const int tx = threadIdx.x, ty = threadIdx.y;
    const int be = blockIdx.x, bd = blockIdx.y;
    const int e = be * 32 + tx;
#pragma unroll
    for (int i = 0; i < 4; ++i) {
        const int d = bd * 32 + ty + 8 * i;
        tile[ty + 8 * i][tx] = P[(size_t)d * E_DIM + e];
    }
    __syncthreads();
    const int dw = bd * 32 + tx;
#pragma unroll
    for (int i = 0; i < 4; ++i) {
        const int ew = be * 32 + ty + 8 * i;
        Pt[(size_t)ew * D_DIM + dw] = f2bf(tile[tx][ty + 8 * i]);
    }
}

// ---------------------------------------------------------------------------
// Kernel: GEMM1  proj[m,e] = sum_d x[m,d]*P[d,e]  (bf16 WMMA / f32 accum)
// Copy-free 2-deep software pipeline: two symmetric register sets, peeled
// epilogue, no conditional register copies (avoids spills/mov chains).
// WMMA A/B use contiguous k = 16h + s in both operands; D layout per CDNA5 ISA 7.12.2.
// ---------------------------------------------------------------------------
union AU { v16bf v; __bf16 b[16]; };

static __device__ __forceinline__ AU load_a_bf16(const float* __restrict__ xrow,
                                                 int d0, int h) {
    AU a;
    const float* lo = xrow + d0 + 16 * h;
    const float* hi = xrow + d0 + 16 * h + 8;
#pragma unroll
    for (int j = 0; j < 8; ++j) {
        a.b[j]     = (__bf16)lo[j];
        a.b[8 + j] = (__bf16)hi[j];
    }
    return a;
}

static __device__ __forceinline__ void load_b4(const unsigned short* __restrict__ Pt,
                                               int e0, int ln, int h, int d0,
                                               v16bf* b) {
#pragma unroll
    for (int t = 0; t < 4; ++t)
        b[t] = *(const v16bf*)(Pt + (size_t)(e0 + 16 * t + ln) * D_DIM + d0 + 16 * h);
}

__global__ void k_gemm1_wmma(const float* __restrict__ x,
                             const unsigned short* __restrict__ Pt,
                             float* __restrict__ proj) {
    const int tid  = threadIdx.x;
    const int lane = tid & 31;
    const int wave = tid >> 5;
    const int h    = lane >> 4;
    const int ln   = lane & 15;
    const int m0   = (blockIdx.x >> 1) * 16;
    const int e0   = ((blockIdx.x & 1) * 4 + wave) * 64;

    const float* __restrict__ xrow = x + (size_t)(m0 + ln) * D_DIM;

    v8f acc[4];
#pragma unroll
    for (int t = 0; t < 4; ++t) acc[t] = v8f_zero();

    AU a0 = load_a_bf16(xrow, 0, h);
    v16bf b0[4];
    load_b4(Pt, e0, ln, h, 0, b0);
    AU a1;
    v16bf b1[4];

#pragma unroll 1
    for (int it = 0; it < D_DIM / 32 - 2; it += 2) {
        a1 = load_a_bf16(xrow, (it + 1) * 32, h);
        load_b4(Pt, e0, ln, h, (it + 1) * 32, b1);
#pragma unroll
        for (int t = 0; t < 4; ++t)
            acc[t] = __builtin_amdgcn_wmma_f32_16x16x32_bf16(
                false, a0.v, false, b0[t], (short)0, acc[t], false, false);
        a0 = load_a_bf16(xrow, (it + 2) * 32, h);
        load_b4(Pt, e0, ln, h, (it + 2) * 32, b0);
#pragma unroll
        for (int t = 0; t < 4; ++t)
            acc[t] = __builtin_amdgcn_wmma_f32_16x16x32_bf16(
                false, a1.v, false, b1[t], (short)0, acc[t], false, false);
    }
    // Epilogue: b0/a0 hold chunk 30; load chunk 31 and consume both.
    a1 = load_a_bf16(xrow, (D_DIM / 32 - 1) * 32, h);
    load_b4(Pt, e0, ln, h, (D_DIM / 32 - 1) * 32, b1);
#pragma unroll
    for (int t = 0; t < 4; ++t)
        acc[t] = __builtin_amdgcn_wmma_f32_16x16x32_bf16(
            false, a0.v, false, b0[t], (short)0, acc[t], false, false);
#pragma unroll
    for (int t = 0; t < 4; ++t)
        acc[t] = __builtin_amdgcn_wmma_f32_16x16x32_bf16(
            false, a1.v, false, b1[t], (short)0, acc[t], false, false);

    // D layout: VGPR r, lane L -> (m = m0 + 8h + r, col = e0 + 16t + (L&15))
#pragma unroll
    for (int t = 0; t < 4; ++t)
#pragma unroll
        for (int r = 0; r < 8; ++r)
            proj[(size_t)(m0 + 8 * h + r) * E_DIM + e0 + 16 * t + ln] = acc[t][r];
}

// ---------------------------------------------------------------------------
// Kernel: fused score GEMM + argmax with async double-buffered LDS staging.
// n-loop unrolled by 2 so the LDS buffer index is compile-time constant.
// LDS panel row stride 1040 B (260 dwords == 4 mod 64) -> conflict-free
// 16-lane ds_load_b128 B reads.
// ---------------------------------------------------------------------------
#define PANEL_STRIDE 1040  // bytes per codeword row in LDS (1024 data + 16 pad)

__global__ void k_score_argmax(const unsigned short* __restrict__ xn,
                               const unsigned short* __restrict__ cn,
                               float* __restrict__ out) {
    __shared__ unsigned char panel[2][16 * PANEL_STRIDE];

    const int tid  = threadIdx.x;
    const int lane = tid & 31;
    const int wave = tid >> 5;
    const int h    = lane >> 4;
    const int ln   = lane & 15;
    const int mt   = blockIdx.x * 4 + wave;   // 0..1023
    const int m0   = mt * 16;

    const unsigned short* __restrict__ xrow = xn + (size_t)(m0 + ln) * E_DIM;

    // Preload full A panel: 16 chunks of K=32 over E=512 (128 VGPRs).
    union AV { v16bf v; v8bf part[2]; };
    AV a[16];
#pragma unroll
    for (int ec = 0; ec < 16; ++ec) {
        a[ec].part[0] = *(const v8bf*)(xrow + 32 * ec + 8 * h);
        a[ec].part[1] = *(const v8bf*)(xrow + 32 * ec + 16 + 8 * h);
    }

    // Cooperative staging mapping: 8 threads per codeword row, 128 B each.
    const unsigned char* cnb = (const unsigned char*)cn;
    const int prow = tid >> 3;            // 0..15: codeword row in panel
    const int pcol = (tid & 7) * 128;     // byte offset within row

    float bestv[8];
    int   besti[8];
#pragma unroll
    for (int r = 0; r < 8; ++r) { bestv[r] = -1e30f; besti[r] = 0; }

    // stage panel for codewords [n0, n0+16) into buffer `buf`
    auto stage = [&](int buf, int n0) {
        const unsigned char* g = cnb + (((size_t)(n0 + prow)) << 10) + pcol;
        unsigned char* l = &panel[buf][prow * PANEL_STRIDE + pcol];
#if HAVE_ASYNC_LDS
#pragma unroll
        for (int j = 0; j < 8; ++j) {
            __builtin_amdgcn_global_load_async_to_lds_b128(
                (as1_v4i*)(g + 16 * j), (as3_v4i*)(l + 16 * j), 0, 0);
        }
#else
#pragma unroll
        for (int j = 0; j < 8; ++j)
            ((float4*)l)[j] = ((const float4*)g)[j];
#endif
    };

    auto sync_stage = [&]() {
#if HAVE_ASYNC_LDS
        wait_async0();
#endif
        __syncthreads();
    };

    // compute 16x16 scores vs codewords [n0, n0+16) from LDS buffer `buf`,
    // update per-lane running argmax
    const int lane_off = ln * PANEL_STRIDE + 16 * h;
    auto consume = [&](int buf, int n0) {
        const unsigned char* base = &panel[buf][lane_off];
        v8f acc = v8f_zero();
#pragma unroll
        for (int ec = 0; ec < 16; ++ec) {
            union { v16bf v; v8bf part[2]; } b;
            b.part[0] = *(const v8bf*)(base + 64 * ec);
            b.part[1] = *(const v8bf*)(base + 64 * ec + 32);
            acc = __builtin_amdgcn_wmma_f32_16x16x32_bf16(
                false, a[ec].v, false, b.v, (short)0, acc, false, false);
        }
        const int idx = n0 + ln;
#pragma unroll
        for (int r = 0; r < 8; ++r) {
            if (acc[r] > bestv[r]) { bestv[r] = acc[r]; besti[r] = idx; }
        }
    };

    stage(0, 0);
    sync_stage();

#pragma unroll 1
    for (int n0 = 0; n0 < K_CB; n0 += 32) {
        stage(1, n0 + 16);
        consume(0, n0);
        sync_stage();
        if (n0 + 32 < K_CB) stage(0, n0 + 32);
        consume(1, n0 + 16);
        sync_stage();
    }

    // Cross-lane argmax within each 16-lane half (rows m0+8h+r live there).
#pragma unroll
    for (int r = 0; r < 8; ++r) {
        float v = bestv[r];
        int   i = besti[r];
#pragma unroll
        for (int off = 8; off >= 1; off >>= 1) {
            float ov = __shfl_xor(v, off, 32);
            int   oi = __shfl_xor(i, off, 32);
            if (ov > v || (ov == v && oi < i)) { v = ov; i = oi; }
        }
        if (ln == 0) out[m0 + 8 * h + r] = (float)i;
    }
}

// ---------------------------------------------------------------------------
// Launch
// ---------------------------------------------------------------------------
extern "C" void kernel_launch(void* const* d_in, const int* in_sizes, int n_in,
                              void* d_out, int out_size, void* d_ws, size_t ws_size,
                              hipStream_t stream) {
    (void)in_sizes; (void)n_in; (void)out_size; (void)ws_size;
    const float* x  = (const float*)d_in[0];   // (B*N, D) = (16384, 1024)
    const float* P  = (const float*)d_in[1];   // (D, E)   = (1024, 512)
    const float* cb = (const float*)d_in[2];   // (K, E)   = (4096, 512)

    unsigned char* ws = (unsigned char*)d_ws;
    const size_t CN_BYTES   = (size_t)K_CB * E_DIM * 2;    //  4 MB
    const size_t PT_BYTES   = (size_t)E_DIM * D_DIM * 2;   //  1 MB
    const size_t PROJ_BYTES = (size_t)M_ROWS * E_DIM * 4;  // 32 MB
    unsigned short* cn   = (unsigned short*)(ws);
    unsigned short* Pt   = (unsigned short*)(ws + CN_BYTES);
    float*          proj = (float*)(ws + CN_BYTES + PT_BYTES);
    unsigned short* xn   = (unsigned short*)(ws + CN_BYTES + PT_BYTES + PROJ_BYTES);
    float* out = (float*)d_out;

    // 1) codebook rows -> unit-norm bf16
    k_rownorm_bf16<<<K_CB, 128, 0, stream>>>(cb, cn);
    // 2) P -> P^T bf16 (contraction-contiguous for WMMA B operand)
    k_transpose_bf16<<<dim3(E_DIM / 32, D_DIM / 32), dim3(32, 8), 0, stream>>>(P, Pt);
    // 3) proj = x @ P   (WMMA bf16 / f32 accum, copy-free pipeline)
    k_gemm1_wmma<<<M_ROWS / 16 * 2, 128, 0, stream>>>(x, Pt, proj);
    // 4) proj rows -> unit-norm bf16
    k_rownorm_bf16<<<M_ROWS, 128, 0, stream>>>(proj, xn);
    // 5) scores + argmax -> indices (as float)
    k_score_argmax<<<M_ROWS / 16 / 4, 128, 0, stream>>>(xn, cn, out);
}
